// DAMSM_20675972563331
// MI455X (gfx1250) — compile-verified, matrix operand order checked by reading, not run on hardware
//
#include <hip/hip_runtime.h>
#include <hip/hip_bf16.h>
#include <cstdint>

// ===========================================================================
// MI455X (gfx1250) DAMSM encoder.
// Convs = implicit GEMM on v_wmma_f32_16x16x32_f16 (f16 in, f32 acc).
// Activations NHWC f16; weights repacked [tap][Cout][Cin] f16 so A/B WMMA
// fragments are contiguous 32B loads. 3x3 conv = 9 shifted 1x1 GEMMs over a
// 10x10 LDS halo tile staged with GLOBAL_LOAD_ASYNC_TO_LDS_B128 (ASYNCcnt),
// double-buffered so DMA of chunk i+1 overlaps WMMA of chunk i.
// Workgroup = 4 waves = 128 Cout x 64 px: each B fragment read from LDS
// feeds 2 WMMAs (1 ds_load_b128 per WMMA). conv1 of each ResBlock fuses
// avgpool2 + add-into-shortcut via a 32KB LDS fp32 tile.
// Attention tail is small fp32 kernels. Workspace requirement: ~285 MB.
// ===========================================================================

typedef __attribute__((ext_vector_type(16))) _Float16 v16h;
typedef __attribute__((ext_vector_type(8)))  _Float16 v8h;
typedef __attribute__((ext_vector_type(8)))  float    v8f;

#define RHO1 4.0f
#define RHO2 5.0f
#define EPS_ 1e-6f

// ---------------------------------------------------------------------------
// Weight repack: OIHW f32 -> [tap][O][I] f16
// ---------------------------------------------------------------------------
__global__ void wcvt_kernel(const float* __restrict__ w, _Float16* __restrict__ o,
                            int O, int I, int T) {
    int idx = blockIdx.x * 256 + threadIdx.x;
    int total = O * I * T;
    if (idx >= total) return;
    int ii = idx % I;
    int rest = idx / I;
    int oo = rest % O;
    int t  = rest / O;
    o[idx] = (_Float16)w[((size_t)oo * I + ii) * T + t];
}

// ---------------------------------------------------------------------------
// Stem: 1x1 conv 3->64, NCHW f32 img -> NHWC f16
// ---------------------------------------------------------------------------
__global__ void stem_kernel(const float* __restrict__ img, const float* __restrict__ w,
                            const float* __restrict__ b, _Float16* __restrict__ out) {
    int p = blockIdx.x * 256 + threadIdx.x;            // pixel over N*H*W
    if (p >= 32 * 128 * 128) return;
    int n = p / (128 * 128);
    int hw = p % (128 * 128);
    const size_t ib = (size_t)n * 3 * 16384 + hw;
    float r0 = img[ib], r1 = img[ib + 16384], r2 = img[ib + 2 * 16384];
    _Float16* o = out + (size_t)p * 64;
    #pragma unroll
    for (int c8 = 0; c8 < 8; ++c8) {
        v8h v;
        #pragma unroll
        for (int j = 0; j < 8; ++j) {
            int c = c8 * 8 + j;
            v[j] = (_Float16)(w[c * 3 + 0] * r0 + w[c * 3 + 1] * r1 + w[c * 3 + 2] * r2 + b[c]);
        }
        *(v8h*)(o + c8 * 8) = v;
    }
}

// ---------------------------------------------------------------------------
// avgpool2 on NHWC f16
// ---------------------------------------------------------------------------
__global__ void pool_kernel(const _Float16* __restrict__ in, _Float16* __restrict__ out,
                            int C, int H, int W) {
    size_t total = (size_t)32 * (H / 2) * (W / 2) * C;
    size_t i = (size_t)blockIdx.x * 256 + threadIdx.x;
    if (i >= total) return;
    int c = i % C;
    size_t q = i / C;
    int pw_ = q % (W / 2); q /= (W / 2);
    int ph  = q % (H / 2);
    int n   = q / (H / 2);
    size_t base = (((size_t)n * H + 2 * ph) * W + 2 * pw_) * C + c;
    size_t rs = (size_t)W * C;
    float s = (float)in[base] + (float)in[base + C] + (float)in[base + rs] + (float)in[base + rs + C];
    out[i] = (_Float16)(0.25f * s);
}

// ---------------------------------------------------------------------------
// Implicit-GEMM conv, WMMA f16. Workgroup = 4 waves = 128 Cout x 64 px (8x8).
//   act : NHWC f16 [32][H][W][CIN]
//   wt  : f16 [KH*KH][COUT][CIN]
//   out : !POOL_ADD -> NHWC f16 [32][H][W][COUT]      (overwrite)
//          POOL_ADD -> NHWC f16 [32][H/2][W/2][COUT]  (+= avgpool2 of result)
// Halo tile staged global->LDS with async DMA, double buffered.
// ---------------------------------------------------------------------------
template <int CIN, int COUT, int H, int W, int KH, bool RELU_IN, bool POOL_ADD>
__global__ __launch_bounds__(128) void conv_wmma(
    const _Float16* __restrict__ act, const _Float16* __restrict__ wt,
    const float* __restrict__ bias, _Float16* __restrict__ out) {
    constexpr int PAD  = (KH == 3) ? 1 : 0;
    constexpr int HALO = 8 + 2 * PAD;                  // 10 (3x3) or 8 (1x1)
    constexpr int TILES_W = W / 8;
    constexpr int STG  = HALO * HALO * 32;             // halves per staging buffer
    constexpr int NCH  = CIN / 32;                     // K chunks

    const int tile = blockIdx.x;
    const int th0  = (tile / TILES_W) * 8;
    const int tw0  = (tile % TILES_W) * 8;
    const int co0  = blockIdx.y * 128;
    const int n    = blockIdx.z;
    const int tidx = threadIdx.x;
    const int wave = tidx >> 5;
    const int lane = tidx & 31;
    const int mrow  = lane & 15;                       // A row / B col / D col
    const int khalf = lane >> 4;                       // which 16-wide K half

    __shared__ float smem[8192];                       // 32KB: 2 halo bufs / pool tile
    _Float16* sIn0 = (_Float16*)smem;                  // [HALO*HALO][32] halves
    _Float16* sIn1 = sIn0 + STG;
    const uint32_t ldsBase0 = (uint32_t)(uintptr_t)(void*)sIn0;
    const uint32_t ldsBase1 = (uint32_t)(uintptr_t)(void*)sIn1;

    v8f acc[2][4] = {{v8f{}, v8f{}, v8f{}, v8f{}}, {v8f{}, v8f{}, v8f{}, v8f{}}};

    // ---- async stage of one 32-channel halo chunk into buffer `which` -----
    auto stage = [&](int ci0, int which) {
        _Float16* dst = which ? sIn1 : sIn0;
        const uint32_t dbase = which ? ldsBase1 : ldsBase0;
        for (int e = tidx; e < HALO * HALO * 4; e += 128) {   // 16B segments
            int p = e >> 2, seg = e & 3;
            int py = p / HALO, px = p % HALO;
            int hy = th0 + py - PAD, wx = tw0 + px - PAD;
            if (hy >= 0 && hy < H && wx >= 0 && wx < W) {
                const _Float16* g = act + ((((size_t)n * H + hy) * W + wx) * CIN + ci0) + seg * 8;
                uint32_t ldsoff = dbase + (uint32_t)(p * 64 + seg * 16);
                asm volatile("global_load_async_to_lds_b128 %0, %1, off"
                             :: "v"(ldsoff), "v"((unsigned long long)(uintptr_t)g)
                             : "memory");
            } else {
                *(v8h*)(dst + p * 32 + seg * 8) = v8h{};      // zero-pad border
            }
        }
    };

    stage(0, 0);
    asm volatile("s_wait_asynccnt 0" ::: "memory");
    __syncthreads();

    for (int ic = 0; ic < NCH; ++ic) {
        const int ci0 = ic * 32;
        _Float16* cur = (ic & 1) ? sIn1 : sIn0;

        if (ic + 1 < NCH) stage((ic + 1) * 32, (ic + 1) & 1); // overlap DMA w/ WMMA

        if (RELU_IN) {                                        // in-place ReLU pass
            for (int e = tidx; e < HALO * HALO * 2; e += 128) {
                v16h v = *(v16h*)(cur + e * 16);
                #pragma unroll
                for (int q = 0; q < 16; ++q) v[q] = v[q] > (_Float16)0 ? v[q] : (_Float16)0;
                *(v16h*)(cur + e * 16) = v;
            }
            __syncthreads();
        }

        // ---- taps: 2 A-fragments x 4 N-subtiles, B reused for 2 WMMAs -----
        #pragma unroll
        for (int t = 0; t < KH * KH; ++t) {
            const int dy = t / KH, dx = t % KH;
            const _Float16* wp0 = wt + (((size_t)t * COUT + (co0 + wave * 32 + mrow)) * CIN + ci0 + khalf * 16);
            const _Float16* wp1 = wp0 + (size_t)16 * CIN;
            const v16h a0 = *(const v16h*)wp0;
            const v16h a1 = *(const v16h*)wp1;
            if (t + 1 < KH * KH)
                __builtin_prefetch((const void*)(wp0 + (size_t)COUT * CIN), 0, 0);
            #pragma unroll
            for (int sub = 0; sub < 4; ++sub) {
                int p  = sub * 16 + mrow;              // pixel column in tile
                int py = (p >> 3) + dy, px = (p & 7) + dx;
                const v16h b = *(const v16h*)(cur + (py * HALO + px) * 32 + khalf * 16);
                acc[0][sub] = __builtin_amdgcn_wmma_f32_16x16x32_f16(
                    false, a0, false, b, (short)0, acc[0][sub], false, false);
                acc[1][sub] = __builtin_amdgcn_wmma_f32_16x16x32_f16(
                    false, a1, false, b, (short)0, acc[1][sub], false, false);
            }
        }

        asm volatile("s_wait_asynccnt 0" ::: "memory");       // next buffer landed
        __syncthreads();
    }

    // ---- bias ------------------------------------------------------------
    float bv[2][8];
    #pragma unroll
    for (int mt = 0; mt < 2; ++mt)
        #pragma unroll
        for (int r = 0; r < 8; ++r)
            bv[mt][r] = bias[co0 + wave * 32 + mt * 16 + khalf * 8 + r];

    if (!POOL_ADD) {
        #pragma unroll
        for (int sub = 0; sub < 4; ++sub) {
            int p  = sub * 16 + mrow;
            int hy = th0 + (p >> 3), wx = tw0 + (p & 7);
            size_t pbase = (((size_t)n * H + hy) * W + wx) * COUT + co0 + wave * 32 + khalf * 8;
            #pragma unroll
            for (int mt = 0; mt < 2; ++mt) {
                v8h o;
                #pragma unroll
                for (int r = 0; r < 8; ++r) o[r] = (_Float16)(acc[mt][sub][r] + bv[mt][r]);
                *(v8h*)(out + pbase + mt * 16) = o;
            }
        }
    } else {
        // fp32 tile [px 64][co 128] in LDS, then 2x2 avgpool += into shortcut
        #pragma unroll
        for (int sub = 0; sub < 4; ++sub) {
            int p = sub * 16 + mrow;
            #pragma unroll
            for (int mt = 0; mt < 2; ++mt)
                #pragma unroll
                for (int r = 0; r < 8; ++r)
                    smem[p * 128 + wave * 32 + mt * 16 + khalf * 8 + r] = acc[mt][sub][r] + bv[mt][r];
        }
        __syncthreads();
        #pragma unroll
        for (int k = 0; k < 16; ++k) {
            int e = tidx + k * 128;                    // 0..2047
            int c = e & 127, q = e >> 7;               // q = pooled pixel 0..15
            int pyq = q >> 2, pxq = q & 3;
            int pA = (2 * pyq) * 8 + 2 * pxq;
            float s = smem[pA * 128 + c] + smem[(pA + 1) * 128 + c] +
                      smem[(pA + 8) * 128 + c] + smem[(pA + 9) * 128 + c];
            size_t oaddr = ((((size_t)n * (H / 2) + (th0 >> 1) + pyq) * (W / 2)) + (tw0 >> 1) + pxq) * COUT + co0 + c;
            out[oaddr] = (_Float16)((float)out[oaddr] + 0.25f * s);
        }
    }
}

// ---------------------------------------------------------------------------
// Attention tail (fp32; ~6 GFLOP total)
// ---------------------------------------------------------------------------
__device__ __forceinline__ float block_reduce(float v, float* red, int tid) {
    red[tid] = v;
    __syncthreads();
    for (int s = 128; s > 0; s >>= 1) {
        if (tid < s) red[tid] += red[tid + s];
        __syncthreads();
    }
    float r = red[0];
    __syncthreads();
    return r;
}

__global__ void rnorm_kernel(const _Float16* __restrict__ pw, float* __restrict__ rnorm) {
    int i = blockIdx.x, r = threadIdx.x;
    const _Float16* col = pw + ((size_t)i * 256 + r) * 256;
    float s = 0.f;
    for (int d = 0; d < 256; ++d) { float v = (float)col[d]; s += v * v; }
    rnorm[i * 256 + r] = sqrtf(s);
}

__global__ void wnorm_kernel(const float* __restrict__ we, float* __restrict__ wn) {
    int bt = blockIdx.x * 256 + threadIdx.x;
    if (bt >= 768) return;
    const float* row = we + (size_t)bt * 256;
    float s = 0.f;
    for (int d = 0; d < 256; ++d) s += row[d] * row[d];
    wn[bt] = sqrtf(s);
}

// attn[i][bt][r] = RHO1 * <we[bt], pw[i][r]> / (wnorm[bt]*rnorm[i][r] + eps)
__global__ void attn_kernel(const float* __restrict__ we, const _Float16* __restrict__ pw,
                            const float* __restrict__ wnorm, const float* __restrict__ rnorm,
                            float* __restrict__ attn) {
    int bt = blockIdx.x, i = blockIdx.y, r = threadIdx.x;
    __shared__ float sw[256];
    sw[r] = we[(size_t)bt * 256 + r];
    __syncthreads();
    const _Float16* col = pw + ((size_t)i * 256 + r) * 256;
    float s = 0.f;
    for (int d = 0; d < 256; ++d) s += sw[d] * (float)col[d];
    float a = RHO1 * s / (wnorm[bt] * rnorm[i * 256 + r] + EPS_);
    attn[((size_t)i * 768 + bt) * 256 + r] = a;
}

// softmax over b (batch axis) for fixed (i,t,r); bt = b*24 + t
__global__ void softmaxb_kernel(float* __restrict__ attn) {
    int idx = blockIdx.x * 256 + threadIdx.x;          // over 32*24*256
    if (idx >= 32 * 24 * 256) return;
    int i = idx / (24 * 256);
    int rem = idx % (24 * 256);
    int t = rem / 256, r = rem % 256;
    size_t base = ((size_t)i * 768 + t) * 256 + r;
    const size_t stride = (size_t)24 * 256;
    float m = -3.4e38f;
    for (int b = 0; b < 32; ++b) { float v = attn[base + b * stride]; m = v > m ? v : m; }
    float s = 0.f;
    for (int b = 0; b < 32; ++b) s += __expf(attn[base + b * stride] - m);
    float inv = 1.f / s;
    for (int b = 0; b < 32; ++b)
        attn[base + b * stride] = __expf(attn[base + b * stride] - m) * inv;
}

// c[bt][d] = sum_r attn[i][bt][r]*region[d][r]; emit sdot=<we[bt],c>, cn2=|c|^2
__global__ void csd_kernel(const float* __restrict__ we, const _Float16* __restrict__ pw,
                           const float* __restrict__ attn, float* __restrict__ sdot,
                           float* __restrict__ cn2) {
    int bt = blockIdx.x, i = blockIdx.y, d = threadIdx.x;
    __shared__ float sa[256];
    __shared__ float red[256];
    sa[d] = attn[((size_t)i * 768 + bt) * 256 + d];
    __syncthreads();
    const _Float16* reg = pw + (size_t)i * 256 * 256;  // [r][d]
    float c = 0.f;
    for (int r = 0; r < 256; ++r) c += sa[r] * (float)reg[(size_t)r * 256 + d];
    float wd = we[(size_t)bt * 256 + d];
    float sd = block_reduce(wd * c, red, d);
    float c2 = block_reduce(c * c, red, d);
    if (d == 0) {
        sdot[(size_t)i * 768 + bt] = sd;
        cn2[(size_t)i * 768 + bt]  = c2;
    }
}

// out[i] = scores_i[i] / sum_b scores_i[b]   (softmax(log(s/rho2)) diagonal)
__global__ void scores_kernel(const float* __restrict__ sdot, const float* __restrict__ cn2,
                              const float* __restrict__ wnorm, float* __restrict__ out) {
    int i = blockIdx.x, t = threadIdx.x;
    __shared__ float red[256];
    float wf = 0.f, cs = 0.f;
    for (int k = t; k < 768; k += 256) {
        wf += wnorm[k] * wnorm[k];
        cs += cn2[(size_t)i * 768 + k];
    }
    float we_fro = sqrtf(block_reduce(wf, red, t));
    float cnorm  = sqrtf(block_reduce(cs, red, t));
    float denom  = we_fro * cnorm + EPS_;
    float tot = 0.f, self = 0.f;
    for (int k = t; k < 768; k += 256) {
        int b = k / 24;
        float e = __expf(RHO2 * sdot[(size_t)i * 768 + k] / denom);
        tot += e;
        if (b == i) self += e;
    }
    tot  = block_reduce(tot, red, t);
    self = block_reduce(self, red, t);
    if (t == 0) out[i] = self / tot;
}

// ===========================================================================
// Launch
// ===========================================================================
extern "C" void kernel_launch(void* const* d_in, const int* in_sizes, int n_in,
                              void* d_out, int out_size, void* d_ws, size_t ws_size,
                              hipStream_t stream) {
    (void)in_sizes; (void)n_in; (void)out_size; (void)ws_size;
    const float* img   = (const float*)d_in[0];
    const float* we    = (const float*)d_in[1];
    const float* rgb_w = (const float*)d_in[2];
    const float* rgb_b = (const float*)d_in[3];
    const float* ws1 = (const float*)d_in[4];  const float* bs1 = (const float*)d_in[5];
    const float* w01 = (const float*)d_in[6];  const float* b01 = (const float*)d_in[7];
    const float* w11 = (const float*)d_in[8];  const float* b11 = (const float*)d_in[9];
    const float* ws2 = (const float*)d_in[10]; const float* bs2 = (const float*)d_in[11];
    const float* w02 = (const float*)d_in[12]; const float* b02 = (const float*)d_in[13];
    const float* w12 = (const float*)d_in[14]; const float* b12 = (const float*)d_in[15];
    const float* ws3 = (const float*)d_in[16]; const float* bs3 = (const float*)d_in[17];
    const float* w03 = (const float*)d_in[18]; const float* b03 = (const float*)d_in[19];
    const float* w13 = (const float*)d_in[20]; const float* b13 = (const float*)d_in[21];
    const float* pk  = (const float*)d_in[22]; const float* pb  = (const float*)d_in[23];

    const size_t MB = 1ull << 20;
    char* base = (char*)d_ws;

    // f16 weight arena @0 (~9.5 MB)
    _Float16* W16 = (_Float16*)base;
    _Float16* WS1 = W16 + 0;        // 128*64
    _Float16* W0_1 = W16 + 8192;    // 9*128*64
    _Float16* W1_1 = W16 + 81920;   // 9*128*128
    _Float16* WS2 = W16 + 229376;   // 256*128
    _Float16* W0_2 = W16 + 262144;  // 9*256*128
    _Float16* W1_2 = W16 + 557056;  // 9*256*256
    _Float16* WS3 = W16 + 1146880;  // 512*256
    _Float16* W0_3 = W16 + 1277952; // 9*512*256
    _Float16* W1_3 = W16 + 2457600; // 9*512*512
    _Float16* WPJ  = W16 + 4816896; // 256*512

    // stats @10MB
    float* rnorm = (float*)(base + 10 * MB);   // 32*256
    float* wnorm = rnorm + 32 * 256;           // 768
    float* sdot  = wnorm + 768;                // 32*768
    float* cn2   = sdot + 32 * 768;            // 32*768

    // activation arenas
    _Float16* X0 = (_Float16*)(base + 16 * MB);   // 64 MB  [32][128][128][64]
    _Float16* T0 = (_Float16*)(base + 80 * MB);   // 128 MB scratch (conv0 out)
    _Float16* Y1 = (_Float16*)(base + 208 * MB);  // 32 MB  [32][64][64][128]
    _Float16* XP = (_Float16*)(base + 240 * MB);  // 16 MB  pooled-input scratch
    _Float16* Y2 = (_Float16*)(base + 256 * MB);  // 16 MB  [32][32][32][256]
    _Float16* Y3 = (_Float16*)(base + 272 * MB);  // 8 MB   [32][16][16][512]
    _Float16* PW = (_Float16*)(base + 280 * MB);  // 4 MB   [32][256][256]
    float*    ATTN = (float*)(base + 16 * MB);    // 25.2 MB (reuses X0 region)

    auto cvt = [&](const float* w, _Float16* o, int O, int I, int T) {
        int total = O * I * T;
        wcvt_kernel<<<(total + 255) / 256, 256, 0, stream>>>(w, o, O, I, T);
    };
    cvt(ws1, WS1, 128, 64, 1);   cvt(w01, W0_1, 128, 64, 9);   cvt(w11, W1_1, 128, 128, 9);
    cvt(ws2, WS2, 256, 128, 1);  cvt(w02, W0_2, 256, 128, 9);  cvt(w12, W1_2, 256, 256, 9);
    cvt(ws3, WS3, 512, 256, 1);  cvt(w03, W0_3, 512, 256, 9);  cvt(w13, W1_3, 512, 512, 9);
    cvt(pk, WPJ, 256, 512, 1);

    auto pool = [&](const _Float16* in, _Float16* o, int C, int H, int W) {
        size_t total = (size_t)32 * (H / 2) * (W / 2) * C;
        pool_kernel<<<(unsigned)((total + 255) / 256), 256, 0, stream>>>(in, o, C, H, W);
    };

    // stem
    stem_kernel<<<2048, 256, 0, stream>>>(img, rgb_w, rgb_b, X0);

    // ResBlock 1: 64 -> 128, 128^2 -> 64^2
    pool(X0, XP, 64, 128, 128);
    conv_wmma<64, 128, 64, 64, 1, false, false><<<dim3(64, 1, 32), 128, 0, stream>>>(XP, WS1, bs1, Y1);
    conv_wmma<64, 128, 128, 128, 3, true, false><<<dim3(256, 1, 32), 128, 0, stream>>>(X0, W0_1, b01, T0);
    conv_wmma<128, 128, 128, 128, 3, true, true><<<dim3(256, 1, 32), 128, 0, stream>>>(T0, W1_1, b11, Y1);

    // ResBlock 2: 128 -> 256, 64^2 -> 32^2
    pool(Y1, XP, 128, 64, 64);
    conv_wmma<128, 256, 32, 32, 1, false, false><<<dim3(16, 2, 32), 128, 0, stream>>>(XP, WS2, bs2, Y2);
    conv_wmma<128, 256, 64, 64, 3, true, false><<<dim3(64, 2, 32), 128, 0, stream>>>(Y1, W0_2, b02, T0);
    conv_wmma<256, 256, 64, 64, 3, true, true><<<dim3(64, 2, 32), 128, 0, stream>>>(T0, W1_2, b12, Y2);

    // ResBlock 3: 256 -> 512, 32^2 -> 16^2
    pool(Y2, XP, 256, 32, 32);
    conv_wmma<256, 512, 16, 16, 1, false, false><<<dim3(4, 4, 32), 128, 0, stream>>>(XP, WS3, bs3, Y3);
    conv_wmma<256, 512, 32, 32, 3, true, false><<<dim3(16, 4, 32), 128, 0, stream>>>(Y2, W0_3, b03, T0);
    conv_wmma<512, 512, 32, 32, 3, true, true><<<dim3(16, 4, 32), 128, 0, stream>>>(T0, W1_3, b13, Y3);

    // projection 512 -> 256 @ 16x16  -> PW[i][r][d]
    conv_wmma<512, 256, 16, 16, 1, false, false><<<dim3(4, 2, 32), 128, 0, stream>>>(Y3, WPJ, pb, PW);

    // attention tail
    rnorm_kernel<<<32, 256, 0, stream>>>(PW, rnorm);
    wnorm_kernel<<<3, 256, 0, stream>>>(we, wnorm);
    attn_kernel<<<dim3(768, 32), 256, 0, stream>>>(we, PW, wnorm, rnorm, ATTN);
    softmaxb_kernel<<<768, 256, 0, stream>>>(ATTN);
    csd_kernel<<<dim3(768, 32), 256, 0, stream>>>(we, PW, ATTN, sdot, cn2);
    scores_kernel<<<32, 256, 0, stream>>>(sdot, cn2, wnorm, (float*)d_out);
}
